// StructureModule_7919919694431
// MI455X (gfx1250) — matrix operand order repaired and checked by reading, not compiled, not run on hardware
//
#include <hip/hip_runtime.h>
#include <hip/hip_bf16.h>
#include <math.h>
#include <stdint.h>

#define NRES 512
#define CS   384
#define CZ   128
#define CIPA 16
#define NH   12
#define NQK  4
#define NV   8
#define NBLOCKS 8
#define CRES 128
#define NANG 7
#define EPSV 1e-8f
#define INFV 100000.0f
#define OUTC 415   // 384 + 14 + 14 + 3

typedef __attribute__((ext_vector_type(16))) _Float16 v16h;
typedef __attribute__((ext_vector_type(8)))  float    v8f;

#define GF_RELU_IN  1
#define GF_RELU_OUT 2

// ---------------------------------------------------------------------------
// LayerNorm: one block per row
// ---------------------------------------------------------------------------
__global__ void k_layernorm(const float* __restrict__ x, const float* __restrict__ g,
                            const float* __restrict__ b, float* __restrict__ y, int cols) {
    int row = blockIdx.x;
    const float* xr = x + (size_t)row * cols;
    float* yr = y + (size_t)row * cols;
    __shared__ float s1[256], s2[256];
    float a = 0.f, a2 = 0.f;
    for (int c = threadIdx.x; c < cols; c += blockDim.x) { float v = xr[c]; a += v; a2 += v * v; }
    s1[threadIdx.x] = a; s2[threadIdx.x] = a2; __syncthreads();
    for (int s = blockDim.x / 2; s > 0; s >>= 1) {
        if (threadIdx.x < (unsigned)s) { s1[threadIdx.x] += s1[threadIdx.x + s]; s2[threadIdx.x] += s2[threadIdx.x + s]; }
        __syncthreads();
    }
    float mu = s1[0] / cols;
    float var = s2[0] / cols - mu * mu;
    float rstd = rsqrtf(var + 1e-5f);
    for (int c = threadIdx.x; c < cols; c += blockDim.x)
        yr[c] = (xr[c] - mu) * rstd * g[c] + b[c];
}

// ---------------------------------------------------------------------------
// Generic WMMA GEMM, compile-time specialized on activation flags.
// 4 waves / block; each wave computes a 16x64 strip (A fragment reused 4x).
// C = act( A[f32 MxK] @ W[f32 KxN] + bias + res ), K % 32 == 0.
// ---------------------------------------------------------------------------
template<int FLAGS>
__global__ void k_gemm_t(const float* __restrict__ A, const float* __restrict__ W,
                         const float* __restrict__ bias, const float* __restrict__ res,
                         float* __restrict__ C, int M, int N, int K) {
    int wave = threadIdx.x >> 5;
    int lane = threadIdx.x & 31;
    int r = lane & 15, hi = lane >> 4;
    int mt = blockIdx.y * 4 + wave;
    int arow = mt * 16 + r;
    int n0 = blockIdx.x * 64;
    v8f acc[4] = {};
    for (int k0 = 0; k0 < K; k0 += 32) {
        v16h af;
        const float* Ap = A + (size_t)arow * K + k0;
        if (k0 + 32 < K) __builtin_prefetch(Ap + 32, 0, 1);
#pragma unroll
        for (int v = 0; v < 4; ++v) {
            float x0 = Ap[hi * 16 + 2 * v];
            float x1 = Ap[hi * 16 + 2 * v + 1];
            float x2 = Ap[8 + hi * 16 + 2 * v];
            float x3 = Ap[8 + hi * 16 + 2 * v + 1];
            if (FLAGS & GF_RELU_IN) {
                x0 = fmaxf(x0, 0.f); x1 = fmaxf(x1, 0.f);
                x2 = fmaxf(x2, 0.f); x3 = fmaxf(x3, 0.f);
            }
            af[2 * v] = (_Float16)x0; af[2 * v + 1] = (_Float16)x1;
            af[8 + 2 * v] = (_Float16)x2; af[8 + 2 * v + 1] = (_Float16)x3;
        }
#pragma unroll
        for (int t = 0; t < 4; ++t) {
            if (n0 + t * 16 < N) {                 // uniform across the wave
                int bcol = n0 + t * 16 + r;
                bool bv = bcol < N;
                v16h bf;
#pragma unroll
                for (int v = 0; v < 8; ++v) {
                    int k = k0 + hi * 16 + 2 * v;
                    float y0 = bv ? W[(size_t)k * N + bcol] : 0.f;
                    float y1 = bv ? W[(size_t)(k + 1) * N + bcol] : 0.f;
                    bf[2 * v] = (_Float16)y0; bf[2 * v + 1] = (_Float16)y1;
                }
                acc[t] = __builtin_amdgcn_wmma_f32_16x16x32_f16(false, af, false, bf,
                                                                (short)0, acc[t], false, false);
            }
        }
    }
#pragma unroll
    for (int t = 0; t < 4; ++t) {
        int bcol = n0 + t * 16 + r;
        if (bcol < N) {
            float bb = bias ? bias[bcol] : 0.f;
#pragma unroll
            for (int q = 0; q < 8; ++q) {
                int orow = mt * 16 + hi * 8 + q;
                float val = acc[t][q] + bb;
                if (res) val += res[(size_t)orow * N + bcol];
                if (FLAGS & GF_RELU_OUT) val = fmaxf(val, 0.f);
                C[(size_t)orow * N + bcol] = val;
            }
        }
    }
}

// ---------------------------------------------------------------------------
// Pair bias precompute (runs once): LN(z) stats + linear(z,b) folded with
// sqrt(1/3) and the square mask.  Stores z mu/rstd for later on-the-fly LN.
// ---------------------------------------------------------------------------
__global__ void k_bias_z(const float* __restrict__ z, const float* __restrict__ gz,
                         const float* __restrict__ bz, const float* __restrict__ Wb,
                         const float* __restrict__ bb, const float* __restrict__ mask,
                         float* __restrict__ bias_pre, float* __restrict__ z_mu,
                         float* __restrict__ z_rstd) {
    int idx = blockIdx.x * blockDim.x + threadIdx.x;   // (i,j) pair
    int i = idx >> 9, j = idx & 511;
    const float* zr = z + (size_t)idx * CZ;
    float a = 0.f, a2 = 0.f;
    for (int c = 0; c < CZ; ++c) { float v = zr[c]; a += v; a2 += v * v; }
    float mu = a / CZ;
    float var = a2 / CZ - mu * mu;
    float rstd = rsqrtf(var + 1e-5f);
    z_mu[idx] = mu; z_rstd[idx] = rstd;
    float acc[NH];
    for (int h = 0; h < NH; ++h) acc[h] = bb[h];
    for (int c = 0; c < CZ; ++c) {
        float v = (zr[c] - mu) * rstd * gz[c] + bz[c];
#pragma unroll
        for (int h = 0; h < NH; ++h) acc[h] += v * Wb[c * NH + h];
    }
    float mterm = INFV * (mask[i] * mask[j] - 1.0f);
    const float s13 = 0.57735026918962576f;
    for (int h = 0; h < NH; ++h)
        bias_pre[(size_t)h * NRES * NRES + (size_t)i * NRES + j] = s13 * acc[h] + mterm;
}

// ---------------------------------------------------------------------------
// Apply rigid frames to raw q/kv points
// ---------------------------------------------------------------------------
__global__ void k_pt_transform(const float* __restrict__ qpraw, const float* __restrict__ kvpraw,
                               const float* __restrict__ rots, const float* __restrict__ trans,
                               float* __restrict__ qp, float* __restrict__ kp, float* __restrict__ vp) {
    int i = blockIdx.x;
    int t = threadIdx.x;     // 0..191
    const float* R = rots + i * 9;
    const float* tr = trans + i * 3;
    if (t < 48) {
        int p = t;
        float lx = qpraw[(size_t)i * 144 + p];
        float ly = qpraw[(size_t)i * 144 + 48 + p];
        float lz = qpraw[(size_t)i * 144 + 96 + p];
        float gx = R[0] * lx + R[1] * ly + R[2] * lz + tr[0];
        float gy = R[3] * lx + R[4] * ly + R[5] * lz + tr[1];
        float gz = R[6] * lx + R[7] * ly + R[8] * lz + tr[2];
        qp[(size_t)i * 144 + p * 3 + 0] = gx;
        qp[(size_t)i * 144 + p * 3 + 1] = gy;
        qp[(size_t)i * 144 + p * 3 + 2] = gz;
    } else {
        int p = t - 48;      // 0..143 = h*12 + pp
        float lx = kvpraw[(size_t)i * 432 + p];
        float ly = kvpraw[(size_t)i * 432 + 144 + p];
        float lz = kvpraw[(size_t)i * 432 + 288 + p];
        float gx = R[0] * lx + R[1] * ly + R[2] * lz + tr[0];
        float gy = R[3] * lx + R[4] * ly + R[5] * lz + tr[1];
        float gz = R[6] * lx + R[7] * ly + R[8] * lz + tr[2];
        int h = p / 12, pp = p % 12;
        if (pp < NQK) {
            size_t o = (size_t)i * 144 + (h * NQK + pp) * 3;
            kp[o + 0] = gx; kp[o + 1] = gy; kp[o + 2] = gz;
        } else {
            size_t o = (size_t)i * 288 + (h * NV + (pp - NQK)) * 3;
            vp[o + 0] = gx; vp[o + 1] = gy; vp[o + 2] = gz;
        }
    }
}

// ---------------------------------------------------------------------------
// Build augmented K=32 fragments for the fused attention-logit GEMM:
//  cols 0..15  : q * 1/sqrt(48)           |  k
//  cols 16..27 : q_pts * hw               |  k_pts
//  col  28     : 1                        |  -0.5*hw*|k_pts|^2
//  col  29     : -0.5*hw*|q_pts|^2        |  1
// ---------------------------------------------------------------------------
__global__ void k_build_aug(const float* __restrict__ q, const float* __restrict__ kv,
                            const float* __restrict__ qp, const float* __restrict__ kp,
                            const float* __restrict__ hw_param,
                            _Float16* __restrict__ Aaug, _Float16* __restrict__ Baug) {
    int i = blockIdx.x;
    int h = threadIdx.x;
    if (h >= NH) return;
    float x = hw_param[h];
    float sp = (x > 0.f) ? x + log1pf(expf(-x)) : log1pf(expf(x));   // softplus
    const float wc  = 0.13608276348795434f;   // sqrt(1/54)
    const float sqk = 0.14433756729740643f;   // 1/sqrt(48)
    float hw = sp * wc;
    _Float16* Ar = Aaug + ((size_t)h * NRES + i) * 32;
    _Float16* Br = Baug + ((size_t)h * NRES + i) * 32;
    for (int c = 0; c < 16; ++c) {
        Ar[c] = (_Float16)(q[(size_t)i * 192 + h * 16 + c] * sqk);
        Br[c] = (_Float16)(kv[(size_t)i * 384 + h * 32 + c]);
    }
    float sq_q = 0.f, sq_k = 0.f;
    for (int d = 0; d < 12; ++d) {
        float qv = qp[(size_t)i * 144 + h * 12 + d];
        float kvv = kp[(size_t)i * 144 + h * 12 + d];
        sq_q += qv * qv; sq_k += kvv * kvv;
        Ar[16 + d] = (_Float16)(qv * hw);
        Br[16 + d] = (_Float16)kvv;
    }
    Ar[28] = (_Float16)1.0f;
    Ar[29] = (_Float16)(-0.5f * hw * sq_q);
    Ar[30] = (_Float16)0.f; Ar[31] = (_Float16)0.f;
    Br[28] = (_Float16)(-0.5f * hw * sq_k);
    Br[29] = (_Float16)1.0f;
    Br[30] = (_Float16)0.f; Br[31] = (_Float16)0.f;
}

// ---------------------------------------------------------------------------
// Attention logits: 4 waves / block, one 16x16 tile per wave,
// single K=32 WMMA + bias_pre add
// ---------------------------------------------------------------------------
__global__ void k_attn_logits(const _Float16* __restrict__ Aaug, const _Float16* __restrict__ Baug,
                              const float* __restrict__ bias_pre, float* __restrict__ aout) {
    int wave = threadIdx.x >> 5;
    int lane = threadIdx.x & 31;
    int jt = blockIdx.x * 4 + wave, it = blockIdx.y, h = blockIdx.z;
    int r = lane & 15, hi = lane >> 4;
    const _Float16* Ap = Aaug + ((size_t)h * NRES + it * 16 + r) * 32 + hi * 16;
    const _Float16* Bp = Baug + ((size_t)h * NRES + jt * 16 + r) * 32 + hi * 16;
    v16h af, bf;
#pragma unroll
    for (int v = 0; v < 4; ++v) {
        af[2 * v] = Ap[2 * v]; af[2 * v + 1] = Ap[2 * v + 1];
        af[8 + 2 * v] = Ap[8 + 2 * v]; af[8 + 2 * v + 1] = Ap[8 + 2 * v + 1];
    }
#pragma unroll
    for (int v = 0; v < 8; ++v) { bf[2 * v] = Bp[2 * v]; bf[2 * v + 1] = Bp[2 * v + 1]; }
    v8f acc = {};
    acc = __builtin_amdgcn_wmma_f32_16x16x32_f16(false, af, false, bf, (short)0, acc, false, false);
    const float* biasr = bias_pre + (size_t)h * NRES * NRES;
    float* outr = aout + (size_t)h * NRES * NRES;
#pragma unroll
    for (int q = 0; q < 8; ++q) {
        int i = it * 16 + hi * 8 + q;
        int j = jt * 16 + r;
        outr[(size_t)i * NRES + j] = acc[q] + biasr[(size_t)i * NRES + j];
    }
}

// ---------------------------------------------------------------------------
// Row softmax (over j), writes f16 probabilities for the WMMA consumers
// ---------------------------------------------------------------------------
__global__ void k_softmax(const float* __restrict__ ain, _Float16* __restrict__ aout) {
    int row = blockIdx.x;           // h*NRES + i
    const float* xr = ain + (size_t)row * NRES;
    _Float16* yr = aout + (size_t)row * NRES;
    __shared__ float red[256];
    float m = -1e30f;
    for (int j = threadIdx.x; j < NRES; j += 256) m = fmaxf(m, xr[j]);
    red[threadIdx.x] = m; __syncthreads();
    for (int s = 128; s > 0; s >>= 1) {
        if (threadIdx.x < (unsigned)s) red[threadIdx.x] = fmaxf(red[threadIdx.x], red[threadIdx.x + s]);
        __syncthreads();
    }
    m = red[0]; __syncthreads();
    float sum = 0.f;
    for (int j = threadIdx.x; j < NRES; j += 256) sum += __expf(xr[j] - m);
    red[threadIdx.x] = sum; __syncthreads();
    for (int s = 128; s > 0; s >>= 1) {
        if (threadIdx.x < (unsigned)s) red[threadIdx.x] += red[threadIdx.x + s];
        __syncthreads();
    }
    float inv = 1.0f / red[0];
    for (int j = threadIdx.x; j < NRES; j += 256) yr[j] = (_Float16)(__expf(xr[j] - m) * inv);
}

// ---------------------------------------------------------------------------
// Pack B = [v (16) | v_pts (24) | zero-pad (8)] per head, f16, ldb = 48
// ---------------------------------------------------------------------------
__global__ void k_build_vB(const float* __restrict__ kv, const float* __restrict__ vp,
                           _Float16* __restrict__ vB) {
    int j = blockIdx.x;
    for (int t = threadIdx.x; t < NH * 48; t += blockDim.x) {
        int h = t / 48, c = t % 48;
        float val = 0.f;
        if (c < 16)       val = kv[(size_t)j * 384 + h * 32 + 16 + c];
        else if (c < 40)  val = vp[(size_t)j * 288 + h * 24 + (c - 16)];
        vB[((size_t)h * NRES + j) * 48 + c] = (_Float16)val;
    }
}

// ---------------------------------------------------------------------------
// o / o_pt GEMM: per head, A = probs (512x512 f16), B = vB (512x48 f16).
// B (48KB) is staged into LDS once per block with async global->LDS DMA
// (GLOBAL_LOAD_ASYNC_TO_LDS_B128, ASYNCcnt), then all 4 waves' WMMAs feed
// their B fragments from LDS.
// ---------------------------------------------------------------------------
__global__ void k_gemm_av(const _Float16* __restrict__ aF, const _Float16* __restrict__ vB,
                          float* __restrict__ oOut) {
    __shared__ _Float16 Bs[NRES * 48];           // 48 KB
    int h = blockIdx.y;
    const _Float16* Bg = vB + (size_t)h * NRES * 48;

    // async-stage B into LDS: 3072 x 16B transfers across 128 threads
    uint32_t bsBase = (uint32_t)(uintptr_t)&Bs[0];
    for (int t = threadIdx.x; t < (NRES * 48 * 2) / 16; t += 128) {
        uint32_t ldsOff = bsBase + (uint32_t)t * 16u;
        uint64_t gptr = (uint64_t)(uintptr_t)Bg + (uint64_t)t * 16u;
        asm volatile("global_load_async_to_lds_b128 %0, %1, off"
                     :: "v"(ldsOff), "v"(gptr) : "memory");
    }
    asm volatile("s_wait_asynccnt 0x0" ::: "memory");
    __syncthreads();

    int wave = threadIdx.x >> 5;
    int lane = threadIdx.x & 31;
    int r = lane & 15, hi = lane >> 4;
    int mt = blockIdx.x * 4 + wave;
    const _Float16* Ab = aF + (size_t)h * NRES * NRES + (size_t)(mt * 16 + r) * NRES;
    v8f acc[3] = {};
    for (int k0 = 0; k0 < NRES; k0 += 32) {
        v16h af;
        const _Float16* Ap = Ab + k0 + hi * 16;
#pragma unroll
        for (int v = 0; v < 4; ++v) {
            af[2 * v] = Ap[2 * v]; af[2 * v + 1] = Ap[2 * v + 1];
            af[8 + 2 * v] = Ap[8 + 2 * v]; af[8 + 2 * v + 1] = Ap[8 + 2 * v + 1];
        }
#pragma unroll
        for (int t = 0; t < 3; ++t) {
            v16h bf;
#pragma unroll
            for (int v = 0; v < 8; ++v) {
                int k = k0 + hi * 16 + 2 * v;
                bf[2 * v]     = Bs[k * 48 + t * 16 + r];
                bf[2 * v + 1] = Bs[(k + 1) * 48 + t * 16 + r];
            }
            acc[t] = __builtin_amdgcn_wmma_f32_16x16x32_f16(false, af, false, bf,
                                                            (short)0, acc[t], false, false);
        }
    }
#pragma unroll
    for (int t = 0; t < 3; ++t)
#pragma unroll
        for (int q = 0; q < 8; ++q) {
            int i = mt * 16 + hi * 8 + q;
            oOut[((size_t)h * NRES + i) * 48 + t * 16 + r] = acc[t][q];
        }
}

// ---------------------------------------------------------------------------
// IPA epilogue: scatter o into cat; inverse-rotate o_pt, compute norms
// cat layout: o(192) | x(96) | y(96) | z(96) | norm(96) | o_pair(1536)
// ---------------------------------------------------------------------------
__global__ void k_ipa_epilogue(const float* __restrict__ oOut, const float* __restrict__ rots,
                               const float* __restrict__ trans, float* __restrict__ cat) {
    int i = blockIdx.x;
    int t = threadIdx.x;   // 0..191
    float* ci = cat + (size_t)i * 2112;
    {
        int h = t >> 4, c = t & 15;
        ci[h * 16 + c] = oOut[((size_t)h * NRES + i) * 48 + c];
    }
    if (t < 96) {
        int h = t >> 3, p = t & 7;
        const float* R = rots + i * 9;
        const float* tr = trans + i * 3;
        const float* g = oOut + ((size_t)h * NRES + i) * 48 + 16 + p * 3;
        float dx = g[0] - tr[0], dy = g[1] - tr[1], dz = g[2] - tr[2];
        float lx = R[0] * dx + R[3] * dy + R[6] * dz;   // R^T * d
        float ly = R[1] * dx + R[4] * dy + R[7] * dz;
        float lz = R[2] * dx + R[5] * dy + R[8] * dz;
        int hp = h * 8 + p;
        ci[192 + hp] = lx; ci[288 + hp] = ly; ci[384 + hp] = lz;
        ci[480 + hp] = sqrtf(lx * lx + ly * ly + lz * lz + EPSV);
    }
}

// ---------------------------------------------------------------------------
// o_pair: per residue i, (16x512 probs) @ (512x128 LN(z)) -> cat[576..2111]
// 8 waves per block, each owning a 16-channel tile; z LN applied on the fly
// from cached mu/rstd.  16 K-step WMMAs per wave + prefetch of next z tile.
// ---------------------------------------------------------------------------
__global__ void k_opair(const _Float16* __restrict__ aF, const float* __restrict__ z,
                        const float* __restrict__ z_mu, const float* __restrict__ z_rstd,
                        const float* __restrict__ gz, const float* __restrict__ bz,
                        float* __restrict__ cat) {
    int i = blockIdx.x;
    int w = threadIdx.x >> 5;       // channel tile 0..7
    int lane = threadIdx.x & 31;
    int r = lane & 15, hi = lane >> 4;
    int ch = w * 16 + r;
    float gc = gz[ch], bc = bz[ch];
    const float* zi  = z + (size_t)i * NRES * CZ;
    const float* mui = z_mu + (size_t)i * NRES;
    const float* rsi = z_rstd + (size_t)i * NRES;
    v8f acc = {};
    for (int k0 = 0; k0 < NRES; k0 += 32) {
        if (k0 + 32 < NRES)
            __builtin_prefetch(zi + (size_t)(k0 + 32 + hi * 16) * CZ + ch, 0, 1);
        v16h af;
#pragma unroll
        for (int e = 0; e < 16; ++e) af[e] = (_Float16)0.f;
        if (r < NH) {
            const _Float16* Ap = aF + (size_t)r * NRES * NRES + (size_t)i * NRES + k0 + hi * 16;
#pragma unroll
            for (int v = 0; v < 4; ++v) {
                af[2 * v] = Ap[2 * v]; af[2 * v + 1] = Ap[2 * v + 1];
                af[8 + 2 * v] = Ap[8 + 2 * v]; af[8 + 2 * v + 1] = Ap[8 + 2 * v + 1];
            }
        }
        v16h bf;
#pragma unroll
        for (int v = 0; v < 8; ++v) {
            int j = k0 + hi * 16 + 2 * v;
            float z0 = (zi[(size_t)j * CZ + ch] - mui[j]) * rsi[j] * gc + bc;
            float z1 = (zi[(size_t)(j + 1) * CZ + ch] - mui[j + 1]) * rsi[j + 1] * gc + bc;
            bf[2 * v] = (_Float16)z0; bf[2 * v + 1] = (_Float16)z1;
        }
        acc = __builtin_amdgcn_wmma_f32_16x16x32_f16(false, af, false, bf, (short)0, acc, false, false);
    }
#pragma unroll
    for (int q = 0; q < 8; ++q) {
        int h = hi * 8 + q;
        if (h < NH) cat[(size_t)i * 2112 + 576 + h * CZ + ch] = acc[q];
    }
}

// ---------------------------------------------------------------------------
// Frame init, backbone update (linear 384->6 + quaternion compose)
// ---------------------------------------------------------------------------
__global__ void k_init_frames(float* __restrict__ rots, float* __restrict__ trans) {
    int i = blockIdx.x * blockDim.x + threadIdx.x;
    if (i < NRES) {
        float* R = rots + i * 9;
        R[0] = 1.f; R[1] = 0.f; R[2] = 0.f;
        R[3] = 0.f; R[4] = 1.f; R[5] = 0.f;
        R[6] = 0.f; R[7] = 0.f; R[8] = 1.f;
        trans[i * 3 + 0] = 0.f; trans[i * 3 + 1] = 0.f; trans[i * 3 + 2] = 0.f;
    }
}

__global__ void k_bb_compose(const float* __restrict__ s, const float* __restrict__ Wbb,
                             const float* __restrict__ bbb, float* __restrict__ rots,
                             float* __restrict__ trans) {
    int i = blockIdx.x * blockDim.x + threadIdx.x;
    if (i >= NRES) return;
    const float* sr = s + (size_t)i * CS;
    float u[6];
    for (int c = 0; c < 6; ++c) {
        float acc = bbb[c];
        for (int k = 0; k < CS; ++k) acc += sr[k] * Wbb[k * 6 + c];
        u[c] = acc;
    }
    float qw = 1.f, qx = u[0], qy = u[1], qz = u[2];
    float n = rsqrtf(qw * qw + qx * qx + qy * qy + qz * qz);
    qw *= n; qx *= n; qy *= n; qz *= n;
    float Ru[9];
    Ru[0] = qw * qw + qx * qx - qy * qy - qz * qz; Ru[1] = 2.f * (qx * qy - qw * qz); Ru[2] = 2.f * (qx * qz + qw * qy);
    Ru[3] = 2.f * (qx * qy + qw * qz); Ru[4] = qw * qw - qx * qx + qy * qy - qz * qz; Ru[5] = 2.f * (qy * qz - qw * qx);
    Ru[6] = 2.f * (qx * qz - qw * qy); Ru[7] = 2.f * (qy * qz + qw * qx); Ru[8] = qw * qw - qx * qx - qy * qy + qz * qz;
    float* R = rots + i * 9;
    float* t = trans + i * 3;
    float Ro[9];
    for (int k = 0; k < 9; ++k) Ro[k] = R[k];
    for (int a = 0; a < 3; ++a)
        for (int b2 = 0; b2 < 3; ++b2)
            R[a * 3 + b2] = Ro[a * 3 + 0] * Ru[b2] + Ro[a * 3 + 1] * Ru[3 + b2] + Ro[a * 3 + 2] * Ru[6 + b2];
    t[0] += Ro[0] * u[3] + Ro[1] * u[4] + Ro[2] * u[5];
    t[1] += Ro[3] * u[3] + Ro[4] * u[4] + Ro[5] * u[5];
    t[2] += Ro[6] * u[3] + Ro[7] * u[4] + Ro[8] * u[5];
}

// ---------------------------------------------------------------------------
// Final output: [s(384) | angles(14) | unnorm(14) | trans*10(3)] per residue
// ---------------------------------------------------------------------------
__global__ void k_final(const float* __restrict__ s, const float* __restrict__ unnorm,
                        const float* __restrict__ trans, float* __restrict__ out) {
    int i = blockIdx.x;
    float* oi = out + (size_t)i * OUTC;
    for (int c = threadIdx.x; c < CS; c += blockDim.x) oi[c] = s[(size_t)i * CS + c];
    if (threadIdx.x < NANG) {
        int t = threadIdx.x;
        float u0 = unnorm[(size_t)i * 14 + 2 * t];
        float u1 = unnorm[(size_t)i * 14 + 2 * t + 1];
        float inv = rsqrtf(fmaxf(u0 * u0 + u1 * u1, EPSV));
        oi[384 + 2 * t] = u0 * inv;
        oi[384 + 2 * t + 1] = u1 * inv;
    }
    if (threadIdx.x < 14) oi[398 + threadIdx.x] = unnorm[(size_t)i * 14 + threadIdx.x];
    if (threadIdx.x < 3)  oi[412 + threadIdx.x] = trans[i * 3 + threadIdx.x] * 10.0f;
}

// ---------------------------------------------------------------------------
// Host orchestration
// ---------------------------------------------------------------------------
extern "C" void kernel_launch(void* const* d_in, const int* in_sizes, int n_in,
                              void* d_out, int out_size, void* d_ws, size_t ws_size,
                              hipStream_t stream) {
    (void)in_sizes; (void)n_in; (void)out_size; (void)ws_size;
    // Inputs (setup_inputs order, params flattened depth-first)
    const float* s_in   = (const float*)d_in[0];
    const float* z_in   = (const float*)d_in[1];
    const float* mask   = (const float*)d_in[2];
    const float* lnS_g  = (const float*)d_in[3];
    const float* lnS_b  = (const float*)d_in[4];
    const float* lnZ_g  = (const float*)d_in[5];
    const float* lnZ_b  = (const float*)d_in[6];
    const float* Win    = (const float*)d_in[7];
    const float* bin    = (const float*)d_in[8];
    const float* Wq     = (const float*)d_in[9];
    const float* bq     = (const float*)d_in[10];
    const float* Wkv    = (const float*)d_in[11];
    const float* bkv    = (const float*)d_in[12];
    const float* Wqp    = (const float*)d_in[13];
    const float* bqp    = (const float*)d_in[14];
    const float* Wkvp   = (const float*)d_in[15];
    const float* bkvp   = (const float*)d_in[16];
    const float* Wb     = (const float*)d_in[17];
    const float* bb_    = (const float*)d_in[18];
    const float* hw_p   = (const float*)d_in[19];
    const float* Wout   = (const float*)d_in[20];
    const float* bout   = (const float*)d_in[21];
    const float* lnI_g  = (const float*)d_in[22];
    const float* lnI_b  = (const float*)d_in[23];
    const float* W1     = (const float*)d_in[24];
    const float* b1     = (const float*)d_in[25];
    const float* W2     = (const float*)d_in[26];
    const float* b2     = (const float*)d_in[27];
    const float* W3     = (const float*)d_in[28];
    const float* b3     = (const float*)d_in[29];
    const float* lnT_g  = (const float*)d_in[30];
    const float* lnT_b  = (const float*)d_in[31];
    const float* Wbb    = (const float*)d_in[32];
    const float* bbb    = (const float*)d_in[33];
    const float* Wa_in  = (const float*)d_in[34];
    const float* ba_in  = (const float*)d_in[35];
    const float* Wa_it  = (const float*)d_in[36];
    const float* ba_it  = (const float*)d_in[37];
    const float* Wblk[2][2] = { { (const float*)d_in[38], (const float*)d_in[40] },
                                { (const float*)d_in[42], (const float*)d_in[44] } };
    const float* bblk[2][2] = { { (const float*)d_in[39], (const float*)d_in[41] },
                                { (const float*)d_in[43], (const float*)d_in[45] } };
    const float* Wa_out = (const float*)d_in[46];
    const float* ba_out = (const float*)d_in[47];

    // Workspace bump allocator (256B aligned)
    uintptr_t wp = (uintptr_t)d_ws;
    auto align256 = [&]() { wp = (wp + 255u) & ~(uintptr_t)255u; };
    auto allocF = [&](size_t n) { align256(); float* p = (float*)wp; wp += n * sizeof(float); return p; };
    auto allocH = [&](size_t n) { align256(); _Float16* p = (_Float16*)wp; wp += n * sizeof(_Float16); return p; };

    float* s_init  = allocF((size_t)NRES * CS);
    float* s_cur   = allocF((size_t)NRES * CS);
    float* s_tmp   = allocF((size_t)NRES * CS);
    float* h1      = allocF((size_t)NRES * CS);
    float* h2      = allocF((size_t)NRES * CS);
    float* cat     = allocF((size_t)NRES * 2112);
    float* qbuf    = allocF((size_t)NRES * 192);
    float* kvbuf   = allocF((size_t)NRES * 384);
    float* qpraw   = allocF((size_t)NRES * 144);
    float* kvpraw  = allocF((size_t)NRES * 432);
    float* qp      = allocF((size_t)NRES * 144);
    float* kp      = allocF((size_t)NRES * 144);
    float* vp      = allocF((size_t)NRES * 288);
    float* bias_pre= allocF((size_t)NH * NRES * NRES);
    float* z_mu    = allocF((size_t)NRES * NRES);
    float* z_rstd  = allocF((size_t)NRES * NRES);
    float* a_f32   = allocF((size_t)NH * NRES * NRES);
    float* oOut    = allocF((size_t)NH * NRES * 48);
    float* rots    = allocF((size_t)NRES * 9);
    float* trans   = allocF((size_t)NRES * 3);
    float* a_ang   = allocF((size_t)NRES * CRES);
    float* h_ang   = allocF((size_t)NRES * CRES);
    float* unnorm  = allocF((size_t)NRES * 14);
    _Float16* Aaug = allocH((size_t)NH * NRES * 32);
    _Float16* Baug = allocH((size_t)NH * NRES * 32);
    _Float16* a_f16= allocH((size_t)NH * NRES * NRES);
    _Float16* vB   = allocH((size_t)NH * NRES * 48);

    auto gemm = [&](const float* A, const float* W, const float* bias, const float* res,
                    float* C, int M, int N, int K, int flags) {
        dim3 grid((N + 63) / 64, M / 64);
        if (flags == GF_RELU_IN)
            k_gemm_t<GF_RELU_IN><<<grid, 128, 0, stream>>>(A, W, bias, res, C, M, N, K);
        else if (flags == GF_RELU_OUT)
            k_gemm_t<GF_RELU_OUT><<<grid, 128, 0, stream>>>(A, W, bias, res, C, M, N, K);
        else
            k_gemm_t<0><<<grid, 128, 0, stream>>>(A, W, bias, res, C, M, N, K);
    };

    // ---- Setup (once) ----
    k_layernorm<<<NRES, 256, 0, stream>>>(s_in, lnS_g, lnS_b, s_init, CS);
    gemm(s_init, Win, bin, nullptr, s_cur, NRES, CS, CS, 0);
    k_bias_z<<<(NRES * NRES) / 256, 256, 0, stream>>>(z_in, lnZ_g, lnZ_b, Wb, bb_, mask,
                                                      bias_pre, z_mu, z_rstd);
    k_init_frames<<<2, 256, 0, stream>>>(rots, trans);

    // ---- 8 sequential IPA blocks ----
    for (int blk = 0; blk < NBLOCKS; ++blk) {
        gemm(s_cur, Wq,   bq,   nullptr, qbuf,   NRES, 192, CS, 0);
        gemm(s_cur, Wkv,  bkv,  nullptr, kvbuf,  NRES, 384, CS, 0);
        gemm(s_cur, Wqp,  bqp,  nullptr, qpraw,  NRES, 144, CS, 0);
        gemm(s_cur, Wkvp, bkvp, nullptr, kvpraw, NRES, 432, CS, 0);
        k_pt_transform<<<NRES, 192, 0, stream>>>(qpraw, kvpraw, rots, trans, qp, kp, vp);
        k_build_aug<<<NRES, 16, 0, stream>>>(qbuf, kvbuf, qp, kp, hw_p, Aaug, Baug);
        k_attn_logits<<<dim3(8, 32, NH), 128, 0, stream>>>(Aaug, Baug, bias_pre, a_f32);
        k_softmax<<<NH * NRES, 256, 0, stream>>>(a_f32, a_f16);
        k_build_vB<<<NRES, 256, 0, stream>>>(kvbuf, vp, vB);
        k_gemm_av<<<dim3(8, NH), 128, 0, stream>>>(a_f16, vB, oOut);
        k_ipa_epilogue<<<NRES, 192, 0, stream>>>(oOut, rots, trans, cat);
        k_opair<<<NRES, 256, 0, stream>>>(a_f16, z_in, z_mu, z_rstd, lnZ_g, lnZ_b, cat);
        gemm(cat, Wout, bout, s_cur, s_tmp, NRES, CS, 2112, 0);
        k_layernorm<<<NRES, 256, 0, stream>>>(s_tmp, lnI_g, lnI_b, s_cur, CS);
        gemm(s_cur, W1, b1, nullptr, h1, NRES, CS, CS, GF_RELU_OUT);
        gemm(h1,    W2, b2, nullptr, h2, NRES, CS, CS, GF_RELU_OUT);
        gemm(h2,    W3, b3, s_cur,  s_tmp, NRES, CS, CS, 0);
        k_layernorm<<<NRES, 256, 0, stream>>>(s_tmp, lnT_g, lnT_b, s_cur, CS);
        k_bb_compose<<<2, 256, 0, stream>>>(s_cur, Wbb, bbb, rots, trans);
    }

    // ---- Angle resnet ----
    gemm(s_cur,  Wa_in, ba_in, nullptr, a_ang, NRES, CRES, CS, GF_RELU_IN);
    gemm(s_init, Wa_it, ba_it, a_ang,   a_ang, NRES, CRES, CS, GF_RELU_IN);
    for (int b = 0; b < 2; ++b) {
        gemm(a_ang, Wblk[b][0], bblk[b][0], nullptr, h_ang, NRES, CRES, CRES, GF_RELU_IN);
        gemm(h_ang, Wblk[b][1], bblk[b][1], a_ang,   a_ang, NRES, CRES, CRES, GF_RELU_IN);
    }
    gemm(a_ang, Wa_out, ba_out, nullptr, unnorm, NRES, 14, CRES, GF_RELU_IN);

    k_final<<<NRES, 256, 0, stream>>>(s_cur, unnorm, trans, (float*)d_out);
}
